// VIT_48722109006375
// MI455X (gfx1250) — matrix operand order repaired and checked by reading, not compiled, not run on hardware
//
#include <hip/hip_runtime.h>
#include <hip/hip_bf16.h>
#include <math.h>

// ---------------- model constants ----------------
#define VDEPTH 12
#define VHEADS 12
#define VDIM   768
#define VHD    64
#define VNTOK  197
#define VB     32
#define VROWS  (VB * VNTOK)           // 6304
#define VSROW  224                    // 197 padded to multiple of 32 (and 16B)
#define VSCALE 0.125f                 // 64^-0.5

#if __has_builtin(__builtin_amdgcn_global_load_async_to_lds_b128) && \
    __has_builtin(__builtin_amdgcn_s_wait_asynccnt)
#define USE_ASYNC_LDS 1
#else
#define USE_ASYNC_LDS 0
#endif

typedef __attribute__((ext_vector_type(16))) _Float16 v16h;
typedef __attribute__((ext_vector_type(8)))  float    v8f;
typedef int vint4 __attribute__((vector_size(4 * sizeof(int))));
#define VAS1 __attribute__((address_space(1)))
#define VAS3 __attribute__((address_space(3)))

__device__ __forceinline__ v16h pack16(float4 p0, float4 p1, float4 p2, float4 p3)
{
    v16h r;
    r[0]  = (_Float16)p0.x; r[1]  = (_Float16)p0.y; r[2]  = (_Float16)p0.z; r[3]  = (_Float16)p0.w;
    r[4]  = (_Float16)p1.x; r[5]  = (_Float16)p1.y; r[6]  = (_Float16)p1.z; r[7]  = (_Float16)p1.w;
    r[8]  = (_Float16)p2.x; r[9]  = (_Float16)p2.y; r[10] = (_Float16)p2.z; r[11] = (_Float16)p2.w;
    r[12] = (_Float16)p3.x; r[13] = (_Float16)p3.y; r[14] = (_Float16)p3.z; r[15] = (_Float16)p3.w;
    return r;
}

// =================================================================
// Generic batched WMMA GEMM:  C = act(scale * A x W^T + bias) [+resid]
//   A[m][k] = A + m*lda + k       W[n][k] = W + n*ldwN + k   (contiguous K)
// Requirements: K % 32 == 0; all row strides multiples of 4 floats.
// M/N edges: load row/col indices clamped (always in-bounds); stores guarded.
// Block = 128 threads (4 waves); waves share one 16-row A tile staged in LDS
// (async global->LDS when available); each wave owns a 16x32 N tile
// (2 accumulators); block tile = 16 x 128.
// =================================================================
__global__ __launch_bounds__(128) void wmma_gemm_kernel(
    const float* __restrict__ A, long long lda, long long sAo, long long sAi,
    const float* __restrict__ W, long long ldwN, long long sWo, long long sWi,
    const float* __restrict__ bias,
    const float* __restrict__ resid, long long ldr,
    float* __restrict__ C, long long ldc, long long sCo, long long sCi,
    int M, int N, int K, int zdiv, float scale, int act)
{
    __shared__ alignas(16) float atile[16 * 32];   // one 16x32 f32 A chunk

    int z  = blockIdx.z;
    int zo = z / zdiv, zi = z % zdiv;
    A += zo * sAo + zi * sAi;
    W += zo * sWo + zi * sWi;
    C += zo * sCo + zi * sCi;

    int lane = threadIdx.x & 31;
    int wave = threadIdx.x >> 5;
    int hi   = lane >> 4;        // lane half
    int l16  = lane & 15;

    int mBase = blockIdx.y * 16;
    int nBase = blockIdx.x * 128 + wave * 32;

    int n0  = nBase + l16;          int n0c = (n0 < N) ? n0 : (N - 1);
    int n1  = nBase + 16 + l16;     int n1c = (n1 < N) ? n1 : (N - 1);

    const float* wPtr0 = W + (long long)n0c * ldwN + hi * 16;
    const float* wPtr1 = W + (long long)n1c * ldwN + hi * 16;

    // A staging assignment: thread t copies one float4 of the 16x32 tile
    int srow = threadIdx.x >> 3;            // 0..15
    int scol = (threadIdx.x & 7) * 4;       // 0,4,...,28
    int srowG = mBase + srow; if (srowG >= M) srowG = M - 1;
    const float* aStage = A + (long long)srowG * lda + scol;
    float* ldsDst = &atile[srow * 32 + scol];
    const float* aFrag = &atile[l16 * 32 + hi * 16];

    v8f acc0 = {}, acc1 = {};

    for (int kt = 0; kt < K; kt += 32) {
        // ---- stage A tile into LDS (async DMA path on gfx1250) ----
#if USE_ASYNC_LDS
        __builtin_amdgcn_global_load_async_to_lds_b128(
            (VAS1 vint4*)(aStage + kt),
            (VAS3 vint4*)ldsDst, 0, 0);
        __builtin_amdgcn_s_wait_asynccnt(0);
#else
        *(float4*)ldsDst = *(const float4*)(aStage + kt);
#endif
        __syncthreads();

        // speculative prefetch of next B tiles (OOB prefetch is dropped)
        __builtin_prefetch(wPtr0 + kt + 32, 0, 0);
        __builtin_prefetch(wPtr1 + kt + 32, 0, 0);

        // ---- A fragment from LDS: af[e]=A[m][hi*16+e] ----
        float4 a0 = *(const float4*)(aFrag);
        float4 a1 = *(const float4*)(aFrag + 4);
        float4 a2 = *(const float4*)(aFrag + 8);
        float4 a3 = *(const float4*)(aFrag + 12);
        v16h af = pack16(a0, a1, a2, a3);

        // ---- B fragments from global: bf[e]=W[n][kt+hi*16+e] ----
        float4 b0 = *(const float4*)(wPtr0 + kt);
        float4 b1 = *(const float4*)(wPtr0 + kt + 4);
        float4 b2 = *(const float4*)(wPtr0 + kt + 8);
        float4 b3 = *(const float4*)(wPtr0 + kt + 12);
        v16h bf0 = pack16(b0, b1, b2, b3);

        float4 c0 = *(const float4*)(wPtr1 + kt);
        float4 c1 = *(const float4*)(wPtr1 + kt + 4);
        float4 c2 = *(const float4*)(wPtr1 + kt + 8);
        float4 c3 = *(const float4*)(wPtr1 + kt + 12);
        v16h bf1 = pack16(c0, c1, c2, c3);

        acc0 = __builtin_amdgcn_wmma_f32_16x16x32_f16(false, af, false, bf0, (short)0, acc0, false, false);
        acc1 = __builtin_amdgcn_wmma_f32_16x16x32_f16(false, af, false, bf1, (short)0, acc1, false, false);

        __syncthreads();   // protect atile before next stage
    }

    // store (C/D layout: VGPR i -> M = i + 8*hi, lane%16 -> N)
    #pragma unroll
    for (int j = 0; j < 2; ++j) {
        int nOut = nBase + j * 16 + l16;
        if (nOut < N) {
            v8f acc = j ? acc1 : acc0;
            #pragma unroll
            for (int i = 0; i < 8; ++i) {
                int mOut = mBase + hi * 8 + i;
                if (mOut < M) {
                    float v = acc[i] * scale;
                    if (bias)  v += bias[nOut];
                    if (act == 1) v = 0.5f * v * (1.0f + erff(v * 0.70710678118654752f));
                    if (resid) v += resid[(long long)mOut * ldr + nOut];
                    C[(long long)mOut * ldc + nOut] = v;
                }
            }
        }
    }
}

// =================================================================
// Direct conv + BN(eval) + ReLU   (stem convs)
// =================================================================
__global__ __launch_bounds__(256) void conv_bn_relu_kernel(
    const float* __restrict__ x, const float* __restrict__ w,
    const float* __restrict__ bnw, const float* __restrict__ bnb,
    const float* __restrict__ bnm, const float* __restrict__ bnv,
    float* __restrict__ out,
    int Cin, int Cout, int Hin, int Win, int Hout, int Wout,
    int ks, int stride, int pad, long long total)
{
    long long idx = (long long)blockIdx.x * blockDim.x + threadIdx.x;
    if (idx >= total) return;
    int ox = (int)(idx % Wout);
    int oy = (int)((idx / Wout) % Hout);
    int oc = (int)((idx / ((long long)Wout * Hout)) % Cout);
    int b  = (int)(idx / ((long long)Wout * Hout * Cout));

    float acc = 0.0f;
    for (int ic = 0; ic < Cin; ++ic) {
        const float* xp = x + ((long long)(b * Cin + ic) * Hin) * Win;
        const float* wp = w + ((long long)(oc * Cin + ic) * ks) * ks;
        for (int kh = 0; kh < ks; ++kh) {
            int iy = oy * stride - pad + kh;
            if (iy < 0 || iy >= Hin) continue;
            for (int kw = 0; kw < ks; ++kw) {
                int ix = ox * stride - pad + kw;
                if (ix < 0 || ix >= Win) continue;
                acc += xp[(long long)iy * Win + ix] * wp[kh * ks + kw];
            }
        }
    }
    float inv = bnw[oc] * rsqrtf(bnv[oc] + 1e-5f);
    float v = (acc - bnm[oc]) * inv + bnb[oc];
    out[idx] = fmaxf(v, 0.0f);
}

// =================================================================
// im2col for the 8x8/s8 patch projection:  [B*196, 64*8*8]
// =================================================================
__global__ __launch_bounds__(256) void im2col_patch_kernel(
    const float* __restrict__ h, float* __restrict__ col, long long total)
{
    long long idx = (long long)blockIdx.x * blockDim.x + threadIdx.x;
    if (idx >= total) return;
    int k  = (int)(idx & 4095);          // ic*64 + kh*8 + kw
    long long m = idx >> 12;             // b*196 + p
    int ic = k >> 6, kh = (k >> 3) & 7, kw = k & 7;
    int p  = (int)(m % 196);
    int b  = (int)(m / 196);
    int py = p / 14, px = p % 14;
    col[idx] = h[(((long long)(b * 64 + ic) * 112) + (py * 8 + kh)) * 112 + (px * 8 + kw)];
}

// =================================================================
// Assemble token sequence: cls + pos_embed, patches + pos_embed
// =================================================================
__global__ __launch_bounds__(256) void assemble_z_kernel(
    const float* __restrict__ ptmp, const float* __restrict__ cls,
    const float* __restrict__ pos, float* __restrict__ z, long long total)
{
    long long idx = (long long)blockIdx.x * blockDim.x + threadIdx.x;
    if (idx >= total) return;
    int c = (int)(idx % VDIM);
    int t = (int)((idx / VDIM) % VNTOK);
    int b = (int)(idx / ((long long)VDIM * VNTOK));
    float v;
    if (t == 0) v = cls[c] + pos[c];
    else        v = ptmp[((long long)(b * 196 + t - 1)) * VDIM + c] + pos[(long long)t * VDIM + c];
    z[idx] = v;
}

// =================================================================
// LayerNorm over `cols` with arbitrary row strides (per-row block)
// =================================================================
__global__ __launch_bounds__(256) void ln_kernel(
    const float* __restrict__ in, long long inStride,
    float* __restrict__ out, long long outStride,
    const float* __restrict__ w, const float* __restrict__ b,
    int cols, float eps)
{
    __shared__ float sS[256], sQ[256];
    long long row = blockIdx.x;
    const float* p = in + row * inStride;
    float s = 0.0f, q = 0.0f;
    for (int c = threadIdx.x; c < cols; c += blockDim.x) {
        float v = p[c]; s += v; q += v * v;
    }
    sS[threadIdx.x] = s; sQ[threadIdx.x] = q;
    __syncthreads();
    for (int off = 128; off > 0; off >>= 1) {
        if ((int)threadIdx.x < off) {
            sS[threadIdx.x] += sS[threadIdx.x + off];
            sQ[threadIdx.x] += sQ[threadIdx.x + off];
        }
        __syncthreads();
    }
    float mean = sS[0] / cols;
    float var  = sQ[0] / cols - mean * mean;
    float inv  = rsqrtf(var + eps);
    float* o = out + row * outStride;
    for (int c = threadIdx.x; c < cols; c += blockDim.x)
        o[c] = (p[c] - mean) * inv * w[c] + b[c];
}

// =================================================================
// Row softmax, one wave per row; row stride may exceed cols (padding)
// =================================================================
__global__ __launch_bounds__(256) void softmax_kernel(
    float* __restrict__ buf, long long rows, long long rowStride, int cols)
{
    int wave = threadIdx.x >> 5;
    int lane = threadIdx.x & 31;
    long long row = (long long)blockIdx.x * (blockDim.x >> 5) + wave;
    if (row >= rows) return;
    float* p = buf + row * rowStride;
    float vals[8];
    int n = 0;
    float mx = -3.4e38f;
    for (int c = lane; c < cols; c += 32) { float v = p[c]; vals[n++] = v; mx = fmaxf(mx, v); }
    for (int off = 16; off > 0; off >>= 1) mx = fmaxf(mx, __shfl_xor(mx, off));
    float s = 0.0f;
    for (int i = 0; i < n; ++i) { float e = __expf(vals[i] - mx); vals[i] = e; s += e; }
    for (int off = 16; off > 0; off >>= 1) s += __shfl_xor(s, off);
    float inv = 1.0f / s;
    n = 0;
    for (int c = lane; c < cols; c += 32) p[c] = vals[n++] * inv;
}

// =================================================================
// ReAttention: 12x12 head mix + bias, then BN over head channel.
// Layout: attn/attn2 [b][h][197][VSROW]; pad cols written as 0.
// =================================================================
__global__ __launch_bounds__(256) void reatt_bn_kernel(
    const float* __restrict__ attn, float* __restrict__ attn2,
    const float* __restrict__ rw, const float* __restrict__ rb,
    const float* __restrict__ bw, const float* __restrict__ bb,
    const float* __restrict__ bm, const float* __restrict__ bv,
    long long total)
{
    long long idx = (long long)blockIdx.x * blockDim.x + threadIdx.x;
    if (idx >= total) return;
    int m = (int)(idx % VSROW);
    long long r = idx / VSROW;
    int n = (int)(r % VNTOK); r /= VNTOK;
    int o = (int)(r % VHEADS);
    long long b = r / VHEADS;
    if (m >= VNTOK) { attn2[idx] = 0.0f; return; }
    const long long HS = (long long)VNTOK * VSROW;
    const float* src = attn + (b * VHEADS) * HS + (long long)n * VSROW + m;
    float acc = rb[o];
    #pragma unroll
    for (int h = 0; h < VHEADS; ++h) acc += rw[o * VHEADS + h] * src[(long long)h * HS];
    float inv = bw[o] * rsqrtf(bv[o] + 1e-5f);
    attn2[idx] = (acc - bm[o]) * inv + bb[o];
}

// =================================================================
// V transpose: vT[b][h][d][m(pad VSROW)] = V[b][m][h][d], zero pad m>=197
// =================================================================
__global__ __launch_bounds__(256) void vtrans_kernel(
    const float* __restrict__ qkv, float* __restrict__ vT, long long total)
{
    long long idx = (long long)blockIdx.x * blockDim.x + threadIdx.x;
    if (idx >= total) return;
    int m = (int)(idx % VSROW);
    long long r = idx / VSROW;
    int d = (int)(r % VHD); r /= VHD;
    int h = (int)(r % VHEADS);
    long long b = r / VHEADS;
    float v = 0.0f;
    if (m < VNTOK)
        v = qkv[((b * VNTOK + m) * 3LL * VDIM) + (2 * VHEADS + h) * VHD + d];
    vT[idx] = v;
}

// =================================================================
// Host-side orchestration
// =================================================================
extern "C" void kernel_launch(void* const* d_in, const int* in_sizes, int n_in,
                              void* d_out, int out_size, void* d_ws, size_t ws_size,
                              hipStream_t stream)
{
    (void)in_sizes; (void)n_in; (void)out_size; (void)ws_size;
    const float* x        = (const float*)d_in[0];
    const float* conv1_w  = (const float*)d_in[1];
    const float* bn1_w    = (const float*)d_in[2];
    const float* bn1_b    = (const float*)d_in[3];
    const float* bn1_m    = (const float*)d_in[4];
    const float* bn1_v    = (const float*)d_in[5];
    const float* conv2_w  = (const float*)d_in[6];
    const float* bn2_w    = (const float*)d_in[7];
    const float* bn2_b    = (const float*)d_in[8];
    const float* bn2_m    = (const float*)d_in[9];
    const float* bn2_v    = (const float*)d_in[10];
    const float* pproj_w  = (const float*)d_in[11];
    const float* pproj_b  = (const float*)d_in[12];
    const float* cls_tok  = (const float*)d_in[13];
    const float* pos_emb  = (const float*)d_in[14];
    const float* ln1_w    = (const float*)d_in[15];
    const float* ln1_b    = (const float*)d_in[16];
    const float* qkv_w    = (const float*)d_in[17];
    const float* reatt_w  = (const float*)d_in[18];
    const float* reatt_b  = (const float*)d_in[19];
    const float* abn_w    = (const float*)d_in[20];
    const float* abn_b    = (const float*)d_in[21];
    const float* abn_m    = (const float*)d_in[22];
    const float* abn_v    = (const float*)d_in[23];
    const float* proj_w   = (const float*)d_in[24];
    const float* proj_b   = (const float*)d_in[25];
    const float* ln2_w    = (const float*)d_in[26];
    const float* ln2_b    = (const float*)d_in[27];
    const float* fc1_w    = (const float*)d_in[28];
    const float* fc1_b    = (const float*)d_in[29];
    const float* fc2_w    = (const float*)d_in[30];
    const float* fc2_b    = (const float*)d_in[31];
    const float* lnf_w    = (const float*)d_in[32];
    const float* lnf_b    = (const float*)d_in[33];
    const float* head_w   = (const float*)d_in[34];
    const float* head_b   = (const float*)d_in[35];

    float* ws = (float*)d_ws;

    // ---- workspace layout (floats), z persistent at 0 ----
    const long long B0    = (long long)VROWS * VDIM;           // 4,841,472
    const long long STEMN = (long long)VB * 64 * 112 * 112;    // 25,690,112
    const long long ATT   = (long long)VB * VHEADS * VNTOK * VSROW; // padded attn
    float* zbuf = ws;
    // stem region (reused by transformer later)
    float* h1   = ws + B0;
    float* h2   = ws + B0 + STEMN;
    float* col  = h1;                        // im2col reuses h1 (exactly fits)
    float* ptmp = ws + B0 + 2 * STEMN;       // [6272,768]
    // transformer region (reuses stem region)
    float* ybuf = ws + B0;
    float* qkv  = ybuf  + (long long)VROWS * VDIM;
    float* attn = qkv   + (long long)VROWS * 3 * VDIM;
    float* attn2= attn  + ATT;
    float* vT   = attn2 + ATT;
    float* aout = vT    + (long long)VB * VHEADS * VHD * VSROW;
    float* hmlp = aout  + (long long)VROWS * VDIM;
    float* clsn = hmlp  + (long long)VROWS * 4 * VDIM;

    auto gemm = [&](const float* A, long long lda, long long sAo, long long sAi,
                    const float* W, long long ldwN, long long sWo, long long sWi,
                    const float* bias, const float* resid, long long ldr,
                    float* C, long long ldc, long long sCo, long long sCi,
                    int M, int N, int K, int zcnt, int zdiv, float scale, int act) {
        dim3 g((N + 127) / 128, (M + 15) / 16, zcnt);
        wmma_gemm_kernel<<<g, 128, 0, stream>>>(A, lda, sAo, sAi,
                                                W, ldwN, sWo, sWi,
                                                bias, resid, ldr,
                                                C, ldc, sCo, sCi,
                                                M, N, K, zdiv, scale, act);
    };

    // ---------------- stem ----------------
    {
        long long t1 = STEMN;
        conv_bn_relu_kernel<<<(int)((t1 + 255) / 256), 256, 0, stream>>>(
            x, conv1_w, bn1_w, bn1_b, bn1_m, bn1_v, h1,
            3, 64, 224, 224, 112, 112, 7, 2, 3, t1);
        conv_bn_relu_kernel<<<(int)((t1 + 255) / 256), 256, 0, stream>>>(
            h1, conv2_w, bn2_w, bn2_b, bn2_m, bn2_v, h2,
            64, 64, 112, 112, 112, 112, 3, 1, 1, t1);
        long long t2 = (long long)VB * 196 * 4096;
        im2col_patch_kernel<<<(int)((t2 + 255) / 256), 256, 0, stream>>>(h2, col, t2);
        // patch projection GEMM: [6272,4096] x [768,4096]^T + b
        gemm(col, 4096, 0, 0, pproj_w, 4096, 0, 0, pproj_b, nullptr, 0,
             ptmp, VDIM, 0, 0, VB * 196, VDIM, 4096, 1, 1, 1.0f, 0);
        long long t3 = (long long)VROWS * VDIM;
        assemble_z_kernel<<<(int)((t3 + 255) / 256), 256, 0, stream>>>(
            ptmp, cls_tok, pos_emb, zbuf, t3);
    }

    // ---------------- transformer blocks ----------------
    const long long qs  = 3LL * VDIM;               // 2304 qkv row stride
    const long long sAb = (long long)VNTOK * qs;    // per-batch stride in qkv rows
    const long long HS  = (long long)VNTOK * VSROW; // per-head attn stride
    for (int d = 0; d < VDEPTH; ++d) {
        // LN1
        ln_kernel<<<VROWS, 256, 0, stream>>>(zbuf, VDIM, ybuf, VDIM,
                                             ln1_w + d * VDIM, ln1_b + d * VDIM, VDIM, 1e-5f);
        // QKV GEMM: [6304,768] x [2304,768]^T
        gemm(ybuf, VDIM, 0, 0, qkv_w + (long long)d * 3 * VDIM * VDIM, VDIM, 0, 0,
             nullptr, nullptr, 0, qkv, qs, 0, 0, VROWS, 3 * VDIM, VDIM, 1, 1, 1.0f, 0);
        // scores: per (b,h)  S = Q K^T * scale   [197,197] (padded ldc)
        gemm(qkv, qs, sAb, VHD,
             qkv + (long long)VHEADS * VHD, qs, sAb, VHD,
             nullptr, nullptr, 0,
             attn, VSROW, (long long)VHEADS * HS, HS,
             VNTOK, VNTOK, VHD, VB * VHEADS, VHEADS, VSCALE, 0);
        // softmax rows
        {
            long long rows = (long long)VB * VHEADS * VNTOK;
            softmax_kernel<<<(int)((rows + 7) / 8), 256, 0, stream>>>(attn, rows, VSROW, VNTOK);
        }
        // re-attention mix + BN (writes zero into pad columns)
        {
            long long tot = ATT;
            reatt_bn_kernel<<<(int)((tot + 255) / 256), 256, 0, stream>>>(
                attn, attn2,
                reatt_w + d * VHEADS * VHEADS, reatt_b + d * VHEADS,
                abn_w + d * VHEADS, abn_b + d * VHEADS,
                abn_m + d * VHEADS, abn_v + d * VHEADS, tot);
        }
        // V transpose with zero pad: vT[b][h][d][m]
        {
            long long tot = (long long)VB * VHEADS * VHD * VSROW;
            vtrans_kernel<<<(int)((tot + 255) / 256), 256, 0, stream>>>(qkv, vT, tot);
        }
        // attn2 x V -> aout[b, n, h*64+d]; K padded to 224 (zeros in vT pad)
        gemm(attn2, VSROW, (long long)VHEADS * HS, HS,
             vT, VSROW, (long long)VHEADS * VHD * VSROW, (long long)VHD * VSROW,
             nullptr, nullptr, 0,
             aout, VDIM, (long long)VNTOK * VDIM, VHD,
             VNTOK, VHD, VSROW, VB * VHEADS, VHEADS, 1.0f, 0);
        // proj GEMM + residual into z
        gemm(aout, VDIM, 0, 0, proj_w + (long long)d * VDIM * VDIM, VDIM, 0, 0,
             proj_b + d * VDIM, zbuf, VDIM,
             zbuf, VDIM, 0, 0, VROWS, VDIM, VDIM, 1, 1, 1.0f, 0);
        // LN2
        ln_kernel<<<VROWS, 256, 0, stream>>>(zbuf, VDIM, ybuf, VDIM,
                                             ln2_w + d * VDIM, ln2_b + d * VDIM, VDIM, 1e-5f);
        // FC1 + exact GELU
        gemm(ybuf, VDIM, 0, 0, fc1_w + (long long)d * 4 * VDIM * VDIM, VDIM, 0, 0,
             fc1_b + (long long)d * 4 * VDIM, nullptr, 0,
             hmlp, 4 * VDIM, 0, 0, VROWS, 4 * VDIM, VDIM, 1, 1, 1.0f, 1);
        // FC2 + residual into z
        gemm(hmlp, 4 * VDIM, 0, 0, fc2_w + (long long)d * VDIM * 4 * VDIM, 4 * VDIM, 0, 0,
             fc2_b + d * VDIM, zbuf, VDIM,
             zbuf, VDIM, 0, 0, VROWS, VDIM, 4 * VDIM, 1, 1, 1.0f, 0);
    }

    // ---------------- head ----------------
    ln_kernel<<<VB, 256, 0, stream>>>(zbuf, (long long)VNTOK * VDIM, clsn, VDIM,
                                      lnf_w, lnf_b, VDIM, 1e-6f);
    gemm(clsn, VDIM, 0, 0, head_w, VDIM, 0, 0, head_b, nullptr, 0,
         (float*)d_out, 1000, 0, 0, VB, 1000, VDIM, 1, 1, 1.0f, 0);
}
